// TemporalGCN_67242007986539
// MI455X (gfx1250) — matrix operand order repaired and checked by reading, not compiled, not run on hardware
//
#include <hip/hip_runtime.h>
#include <hip/hip_bf16.h>
#include <cstdint>

#define NNODES 50000
#define NEDGES 800000
#define FDIM   8
#define PDIM   12
#define HD     256
#define HID    128
#define ODIM   12

typedef __attribute__((ext_vector_type(16))) _Float16 v16h;
typedef __attribute__((ext_vector_type(8)))  _Float16 v8h;
typedef __attribute__((ext_vector_type(8)))  float    v8f;
typedef __attribute__((ext_vector_type(4)))  int      i4vec;

#if defined(__AMDGCN__) && __has_builtin(__builtin_amdgcn_global_load_async_to_lds_b128)
#define ASYNC_LDS 1
#else
#define ASYNC_LDS 0
#endif

#if ASYNC_LDS
typedef __attribute__((address_space(1))) i4vec* gptr128;   // global 128-bit src
typedef __attribute__((address_space(3))) i4vec* lptr128;   // LDS 128-bit dst
__device__ __forceinline__ gptr128 to_global(const void* p) {
    return (gptr128)(uintptr_t)p;
}
__device__ __forceinline__ lptr128 to_lds(void* p) {
    return (lptr128)(uintptr_t)p;   // generic LDS addr low 32 bits == LDS offset
}
#endif

// ---------------------------------------------------------------------------
// LDS-staged WMMA GEMM: D[nrows,NC] = act(A[nrows,K](f16) @ B[K,NC] + bias),
// B given as BT[NC,K] f16 row-major.
// Block: 256 threads (8 waves), block tile 128(M) x 64(N), K chunk 32.
// A/B chunks staged into LDS with async global->LDS copies (ASYNCcnt) when
// available, else load+ds_store. Each wave: one 16-row M-tile x four 16-col
// N-tiles -> 4 f32 accumulators, 4 x v_wmma_f32_16x16x32_f16 per K chunk with
// all B fragments preloaded so the WMMAs issue back-to-back.
// Fragment layouts per CDNA5 ISA:
//   A 16x32 f16: lane-half hi holds K = k+8*hi..+7 and K = k+16+8*hi..+7
//   B 32x16 f16: lane-half hi holds K = k+8*hi..+7 and K = k+16+8*hi..+7
//   C/D f32:     VGPR r, lane l -> M = r + 8*(l>=16), N = l&15
// ---------------------------------------------------------------------------
template<int K, int NC, int ACT, bool BIAS>
__global__ __launch_bounds__(256) void wmma_gemm_lds_kernel(
    const _Float16* __restrict__ A, const _Float16* __restrict__ BT,
    const float* __restrict__ bias, float* __restrict__ D, int nrows)
{
    constexpr int BM = 128, BN = 64, BK = 32;
    __shared__ _Float16 As[BM * BK];   // [row][k] 8 KB
    __shared__ _Float16 Bs[BN * BK];   // [col][k] 4 KB

    const int tid  = threadIdx.x;
    const int lane = tid & 31;
    const int wave = tid >> 5;
    const int nBlkN = NC / BN;
    const int blkM = blockIdx.x / nBlkN;
    const int blkN = blockIdx.x % nBlkN;
    const int rowBase = blkM * BM;
    const int colBase = blkN * BN;

    const int lr = lane & 15;
    const int hi = lane >> 4;

    // cooperative-load coordinates
    const int ar = tid >> 1;              // A: 2 threads per row, 32B each
    const int ak = (tid & 1) * 16;
    const int bc = tid >> 2;              // B: 4 threads per col, 16B each
    const int bk = (tid & 3) * 8;
    const bool arow_ok = (rowBase + ar) < nrows;
    const _Float16* __restrict__ agp = A  + (size_t)(rowBase + ar) * K + ak;
    const _Float16* __restrict__ bgp = BT + (size_t)(colBase + bc) * K + bk;

    v8f acc[4] = {};

    for (int k = 0; k < K; k += BK) {
        // ---- stage A (128x32) and B (64x32) chunks into LDS ----
#if ASYNC_LDS
        if (arow_ok) {
            __builtin_amdgcn_global_load_async_to_lds_b128(
                to_global(agp + k),     to_lds(As + ar * BK + ak),     0, 0);
            __builtin_amdgcn_global_load_async_to_lds_b128(
                to_global(agp + k + 8), to_lds(As + ar * BK + ak + 8), 0, 0);
        } else {
            v8h z = {};
            *(v8h*)(As + ar * BK + ak)     = z;
            *(v8h*)(As + ar * BK + ak + 8) = z;
        }
        __builtin_amdgcn_global_load_async_to_lds_b128(
            to_global(bgp + k), to_lds(Bs + bc * BK + bk), 0, 0);
#else
        v8h a0 = {}, a1 = {};
        if (arow_ok) {
            a0 = *(const v8h*)(agp + k);
            a1 = *(const v8h*)(agp + k + 8);
        }
        *(v8h*)(As + ar * BK + ak)     = a0;
        *(v8h*)(As + ar * BK + ak + 8) = a1;
        *(v8h*)(Bs + bc * BK + bk)     = *(const v8h*)(bgp + k);
#endif
        if (k + BK < K) {   // prefetch next chunk toward L2/L0
            __builtin_prefetch(agp + k + BK, 0, 3);
            __builtin_prefetch(bgp + k + BK, 0, 3);
        }
#if ASYNC_LDS
#if __has_builtin(__builtin_amdgcn_s_wait_asynccnt)
        __builtin_amdgcn_s_wait_asynccnt(0);
#else
        asm volatile("s_wait_asynccnt 0" ::: "memory");
#endif
#endif
        __syncthreads();

        // ---- A fragment for this wave's M-tile (from LDS) ----
        const int arow = wave * 16 + lr;
        v16h afrag;
        *((v8h*)&afrag + 0) = *(const v8h*)(As + arow * BK + hi * 8);
        *((v8h*)&afrag + 1) = *(const v8h*)(As + arow * BK + 16 + hi * 8);

        // ---- preload all 4 B fragments, then 4 WMMAs back-to-back ----
        v16h bfrag[4];
#pragma unroll
        for (int tn = 0; tn < 4; ++tn) {
            const int bcol = tn * 16 + lr;
            *((v8h*)&bfrag[tn] + 0) = *(const v8h*)(Bs + bcol * BK + hi * 8);
            *((v8h*)&bfrag[tn] + 1) = *(const v8h*)(Bs + bcol * BK + 16 + hi * 8);
        }
#pragma unroll
        for (int tn = 0; tn < 4; ++tn) {
            acc[tn] = __builtin_amdgcn_wmma_f32_16x16x32_f16(
                false, afrag, false, bfrag[tn], (short)0, acc[tn], false, false);
        }
        __syncthreads();
    }

    // ---- epilogue: bias + activation + store ----
#pragma unroll
    for (int tn = 0; tn < 4; ++tn) {
        const int col = colBase + tn * 16 + lr;
        const float bv = BIAS ? bias[col] : 0.0f;
#pragma unroll
        for (int r = 0; r < 8; ++r) {
            const int orow = rowBase + wave * 16 + r + hi * 8;
            if (orow >= nrows) continue;
            float v = acc[tn][r] + bv;
            if (ACT == 1) v = 1.0f / (1.0f + __expf(-v));       // sigmoid
            else if (ACT == 2) v = tanhf(v);                    // tanh
            else if (ACT == 3) v = fmaxf(v, 0.0f);              // relu
            D[(size_t)orow * NC + col] = v;
        }
    }
}

static inline int gemm_grid(int nrows, int nc) {
    return ((nrows + 127) / 128) * (nc / 64);
}

// ---------------------------------------------------------------------------
// Graph normalization precompute
// ---------------------------------------------------------------------------
__global__ void deg_accum_kernel(const long long* __restrict__ src,
                                 const long long* __restrict__ dst,
                                 const float* __restrict__ w,
                                 float* __restrict__ deg_g, float* __restrict__ deg_c)
{
    int e = blockIdx.x * blockDim.x + threadIdx.x;
    if (e >= NEDGES) return;
    float we = w[e];
    atomicAdd(deg_g + dst[e], we);
    atomicAdd(deg_c + src[e], we);
}

__global__ void node_norm_kernel(const float* __restrict__ deg_g,
                                 const float* __restrict__ deg_c,
                                 float* __restrict__ dinv_g, float* __restrict__ dinv_c,
                                 float* __restrict__ self_g)
{
    int i = blockIdx.x * blockDim.x + threadIdx.x;
    if (i >= NNODES) return;
    float dg = deg_g[i] + 1.0f;                     // GCN adds self loop (fill 1.0)
    float ig = rsqrtf(fmaxf(dg, 1e-12f));
    dinv_g[i] = ig;
    self_g[i] = ig * ig;
    float dc = deg_c[i];
    dinv_c[i] = (dc > 0.0f) ? rsqrtf(fmaxf(dc, 1e-12f)) : 0.0f;
}

__global__ void edge_norm_kernel(const long long* __restrict__ src,
                                 const long long* __restrict__ dst,
                                 const float* __restrict__ w,
                                 const float* __restrict__ dinv_g,
                                 const float* __restrict__ dinv_c,
                                 float* __restrict__ norm_g, float* __restrict__ norm_c)
{
    int e = blockIdx.x * blockDim.x + threadIdx.x;
    if (e >= NEDGES) return;
    long long s = src[e], d = dst[e];
    float we = w[e];
    norm_g[e] = dinv_g[s] * we * dinv_g[d];
    norm_c[e] = -(dinv_c[s] * we * dinv_c[d]);
}

__global__ void softmax_kernel(const float* __restrict__ att, float* __restrict__ probs)
{
    if (threadIdx.x == 0 && blockIdx.x == 0) {
        float m = -1e30f;
        for (int i = 0; i < PDIM; ++i) m = fmaxf(m, att[i]);
        float e[PDIM], s = 0.0f;
        for (int i = 0; i < PDIM; ++i) { e[i] = __expf(att[i] - m); s += e[i]; }
        float inv = 1.0f / s;
        for (int i = 0; i < PDIM; ++i) probs[i] = e[i] * inv;
    }
}

// W[K][NC] (f32) -> BT[NC][K] (f16)
__global__ void transpose_f16_kernel(const float* __restrict__ W,
                                     _Float16* __restrict__ BT, int K, int NC)
{
    int idx = blockIdx.x * blockDim.x + threadIdx.x;
    if (idx >= K * NC) return;
    int k = idx / NC, c = idx % NC;
    BT[(size_t)c * K + k] = (_Float16)W[idx];
}

// ---------------------------------------------------------------------------
// Per-timestep kernels
// ---------------------------------------------------------------------------
// ChebConv T1 propagation over F=8 features (x is [N, F, P] strided)
__global__ void cheb_scatter_kernel(const long long* __restrict__ src,
                                    const long long* __restrict__ dst,
                                    const float* __restrict__ norm_c,
                                    const float* __restrict__ x, int p,
                                    float* __restrict__ tx1)
{
    int e = blockIdx.x * blockDim.x + threadIdx.x;
    if (e >= NEDGES) return;
    float w = norm_c[e];
    long long s = src[e], d = dst[e];
#pragma unroll
    for (int f = 0; f < FDIM; ++f) {
        float xv = x[(size_t)s * FDIM * PDIM + f * PDIM + p];
        atomicAdd(tx1 + (size_t)d * FDIM + f, w * xv);
    }
}

// h = xp@W0 + tx1@W1 + cb ; xz = xp@Wz ; xr = xp@Wr ; xh = xp@Wh
__global__ __launch_bounds__(256) void node_transform_kernel(
    const float* __restrict__ x, int p, const float* __restrict__ tx1,
    const float* __restrict__ W0, const float* __restrict__ W1,
    const float* __restrict__ cb,
    const float* __restrict__ Wz, const float* __restrict__ Wr,
    const float* __restrict__ Wh,
    float* __restrict__ h, float* __restrict__ xz,
    float* __restrict__ xr, float* __restrict__ xh)
{
    int idx = blockIdx.x * blockDim.x + threadIdx.x;
    if (idx >= NNODES * HD) return;
    int node = idx >> 8, c = idx & (HD - 1);
    float hv = cb[c], zv = 0.0f, rv = 0.0f, hh = 0.0f;
#pragma unroll
    for (int f = 0; f < FDIM; ++f) {
        float xv = x[(size_t)node * FDIM * PDIM + f * PDIM + p];
        float tv = tx1[(size_t)node * FDIM + f];
        hv = fmaf(xv, W0[f * HD + c], hv);
        hv = fmaf(tv, W1[f * HD + c], hv);
        zv = fmaf(xv, Wz[f * HD + c], zv);
        rv = fmaf(xv, Wr[f * HD + c], rv);
        hh = fmaf(xv, Wh[f * HD + c], hh);
    }
    h[idx] = hv; xz[idx] = zv; xr[idx] = rv; xh[idx] = hh;
}

// scatter-add over edges, HD=256 feats, 4 floats per thread
__global__ __launch_bounds__(256) void gcn_scatter_kernel(
    const long long* __restrict__ src, const long long* __restrict__ dst,
    const float* __restrict__ norm_g, const float* __restrict__ xw,
    float* __restrict__ out)
{
    long long idx = (long long)blockIdx.x * blockDim.x + threadIdx.x;
    int e = (int)(idx >> 6);
    if (e >= NEDGES) return;
    int f = ((int)idx & 63) << 2;
    float w = norm_g[e];
    long long s = src[e], d = dst[e];
    const float4 v = *(const float4*)(xw + (size_t)s * HD + f);
    float* o = out + (size_t)d * HD + f;
    atomicAdd(o + 0, w * v.x);
    atomicAdd(o + 1, w * v.y);
    atomicAdd(o + 2, w * v.z);
    atomicAdd(o + 3, w * v.w);
}

// c += self_g * xw + bias, fused across the three gates
__global__ __launch_bounds__(256) void gcn_self_kernel(
    const float* __restrict__ self_g,
    const float* __restrict__ xz, const float* __restrict__ xr,
    const float* __restrict__ xh,
    const float* __restrict__ bz, const float* __restrict__ br,
    const float* __restrict__ bh,
    float* __restrict__ cz, float* __restrict__ cr, float* __restrict__ ch)
{
    int idx = blockIdx.x * blockDim.x + threadIdx.x;
    if (idx >= NNODES * HD) return;
    int node = idx >> 8, c = idx & (HD - 1);
    float sg = self_g[node];
    cz[idx] += sg * xz[idx] + bz[c];
    cr[idx] += sg * xr[idx] + br[c];
    ch[idx] += sg * xh[idx] + bh[c];
}

// Acat[n, 0:256] = a[n], Acat[n, 256:512] = b[n]  (f32 -> f16)
__global__ __launch_bounds__(256) void pack2_kernel(
    const float* __restrict__ a, const float* __restrict__ b,
    _Float16* __restrict__ out)
{
    long long idx = (long long)blockIdx.x * blockDim.x + threadIdx.x;
    if (idx >= (long long)NNODES * 2 * HD) return;
    int node = (int)(idx >> 9), j = (int)idx & 511;
    float v = (j < HD) ? a[(size_t)node * HD + j] : b[(size_t)node * HD + (j - HD)];
    out[idx] = (_Float16)v;
}

// Acat[n, 0:256] = a[n], Acat[n, 256:512] = h[n]*R[n]
__global__ __launch_bounds__(256) void pack2_mul_kernel(
    const float* __restrict__ a, const float* __restrict__ h,
    const float* __restrict__ R, _Float16* __restrict__ out)
{
    long long idx = (long long)blockIdx.x * blockDim.x + threadIdx.x;
    if (idx >= (long long)NNODES * 2 * HD) return;
    int node = (int)(idx >> 9), j = (int)idx & 511;
    float v;
    if (j < HD) v = a[(size_t)node * HD + j];
    else { int k = j - HD; v = h[(size_t)node * HD + k] * R[(size_t)node * HD + k]; }
    out[idx] = (_Float16)v;
}

// Hacc += prob[p] * (Z*h + (1-Z)*Ht)
__global__ __launch_bounds__(256) void gru_update_kernel(
    const float* __restrict__ Z, const float* __restrict__ h,
    const float* __restrict__ Ht, const float* __restrict__ probs, int p,
    float* __restrict__ Hacc)
{
    int idx = blockIdx.x * blockDim.x + threadIdx.x;
    if (idx >= NNODES * HD) return;
    float pr = probs[p];
    float z = Z[idx];
    Hacc[idx] += pr * (z * h[idx] + (1.0f - z) * Ht[idx]);
}

__global__ __launch_bounds__(256) void relu_pack_kernel(
    const float* __restrict__ Hacc, _Float16* __restrict__ Hr)
{
    int idx = blockIdx.x * blockDim.x + threadIdx.x;
    if (idx >= NNODES * HD) return;
    Hr[idx] = (_Float16)fmaxf(Hacc[idx], 0.0f);
}

// out[n, 0:12] = h1[n, 0:128] @ W2[128,12] + b2
__global__ __launch_bounds__(256) void head2_kernel(
    const float* __restrict__ h1, const float* __restrict__ W2,
    const float* __restrict__ b2, float* __restrict__ out)
{
    int idx = blockIdx.x * blockDim.x + threadIdx.x;
    if (idx >= NNODES * ODIM) return;
    int node = idx / ODIM, o = idx % ODIM;
    float acc = b2[o];
    const float* hr = h1 + (size_t)node * HID;
#pragma unroll 4
    for (int k = 0; k < HID; ++k) acc = fmaf(hr[k], W2[k * ODIM + o], acc);
    out[idx] = acc;
}

// ---------------------------------------------------------------------------
extern "C" void kernel_launch(void* const* d_in, const int* in_sizes, int n_in,
                              void* d_out, int out_size, void* d_ws, size_t ws_size,
                              hipStream_t stream) {
    const float*     x        = (const float*)d_in[0];
    const long long* eidx     = (const long long*)d_in[1];   // int64 [2,E]
    const float*     eattr    = (const float*)d_in[2];
    const float*     att      = (const float*)d_in[3];
    const float*     cheb_W0  = (const float*)d_in[4];
    const float*     cheb_W1  = (const float*)d_in[5];
    const float*     cheb_b   = (const float*)d_in[6];
    const float*     Wz = (const float*)d_in[7],  *bz = (const float*)d_in[8];
    const float*     Lz = (const float*)d_in[9],  *blz = (const float*)d_in[10];
    const float*     Wr = (const float*)d_in[11], *br = (const float*)d_in[12];
    const float*     Lr = (const float*)d_in[13], *blr = (const float*)d_in[14];
    const float*     Wh = (const float*)d_in[15], *bh = (const float*)d_in[16];
    const float*     Lh = (const float*)d_in[17], *blh = (const float*)d_in[18];
    const float*     W1 = (const float*)d_in[19], *b1 = (const float*)d_in[20];
    const float*     W2 = (const float*)d_in[21], *b2 = (const float*)d_in[22];
    float* out = (float*)d_out;   // [N*12] then [N*256]

    const long long* src = eidx;
    const long long* dst = eidx + NEDGES;

    // ---- workspace carve-up (256B aligned) ----
    char* wp = (char*)d_ws;
    auto alloc = [&](size_t bytes) -> void* {
        void* r = (void*)wp;
        wp += (bytes + 255) & ~(size_t)255;
        return r;
    };
    float* deg_g  = (float*)alloc(NNODES * 4);
    float* deg_c  = (float*)alloc(NNODES * 4);
    float* dinv_g = (float*)alloc(NNODES * 4);
    float* dinv_c = (float*)alloc(NNODES * 4);
    float* self_g = (float*)alloc(NNODES * 4);
    float* norm_g = (float*)alloc((size_t)NEDGES * 4);
    float* norm_c = (float*)alloc((size_t)NEDGES * 4);
    float* probs  = (float*)alloc(16 * 4);
    float* tx1    = (float*)alloc((size_t)NNODES * FDIM * 4);
    float* hbuf   = (float*)alloc((size_t)NNODES * HD * 4);
    float* xz     = (float*)alloc((size_t)NNODES * HD * 4);   // reused as Z
    float* xr     = (float*)alloc((size_t)NNODES * HD * 4);   // reused as R
    float* xh     = (float*)alloc((size_t)NNODES * HD * 4);   // reused as Ht
    float* cz     = (float*)alloc((size_t)NNODES * HD * 4);
    float* cr     = (float*)alloc((size_t)NNODES * HD * 4);
    float* ch     = (float*)alloc((size_t)NNODES * HD * 4);
    float* Hacc   = (float*)alloc((size_t)NNODES * HD * 4);
    float* h1     = (float*)alloc((size_t)NNODES * HID * 4);
    _Float16* Acat = (_Float16*)alloc((size_t)NNODES * 2 * HD * 2);
    _Float16* Hr   = (_Float16*)alloc((size_t)NNODES * HD * 2);
    _Float16* LzT  = (_Float16*)alloc((size_t)2 * HD * HD * 2);
    _Float16* LrT  = (_Float16*)alloc((size_t)2 * HD * HD * 2);
    _Float16* LhT  = (_Float16*)alloc((size_t)2 * HD * HD * 2);
    _Float16* W1T  = (_Float16*)alloc((size_t)HD * HID * 2);

    const int TB = 256;
    const int gE    = (NEDGES + TB - 1) / TB;
    const int gN    = (NNODES + TB - 1) / TB;
    const int gNH   = (NNODES * HD + TB - 1) / TB;
    const int gPack = (int)(((long long)NNODES * 2 * HD + TB - 1) / TB);
    const int gScat = (int)(((long long)NEDGES * 64 + TB - 1) / TB);

    // ---- graph normalization ----
    (void)hipMemsetAsync(deg_g, 0, NNODES * 4, stream);
    (void)hipMemsetAsync(deg_c, 0, NNODES * 4, stream);
    deg_accum_kernel<<<gE, TB, 0, stream>>>(src, dst, eattr, deg_g, deg_c);
    node_norm_kernel<<<gN, TB, 0, stream>>>(deg_g, deg_c, dinv_g, dinv_c, self_g);
    edge_norm_kernel<<<gE, TB, 0, stream>>>(src, dst, eattr, dinv_g, dinv_c, norm_g, norm_c);
    softmax_kernel<<<1, 32, 0, stream>>>(att, probs);

    // ---- weights -> f16 transposed ----
    transpose_f16_kernel<<<(2 * HD * HD + TB - 1) / TB, TB, 0, stream>>>(Lz, LzT, 2 * HD, HD);
    transpose_f16_kernel<<<(2 * HD * HD + TB - 1) / TB, TB, 0, stream>>>(Lr, LrT, 2 * HD, HD);
    transpose_f16_kernel<<<(2 * HD * HD + TB - 1) / TB, TB, 0, stream>>>(Lh, LhT, 2 * HD, HD);
    transpose_f16_kernel<<<(HD * HID + TB - 1) / TB, TB, 0, stream>>>(W1, W1T, HD, HID);

    (void)hipMemsetAsync(Hacc, 0, (size_t)NNODES * HD * 4, stream);

    // ---- temporal loop ----
    for (int p = 0; p < PDIM; ++p) {
        (void)hipMemsetAsync(tx1, 0, (size_t)NNODES * FDIM * 4, stream);
        (void)hipMemsetAsync(cz, 0, (size_t)NNODES * HD * 4, stream);
        (void)hipMemsetAsync(cr, 0, (size_t)NNODES * HD * 4, stream);
        (void)hipMemsetAsync(ch, 0, (size_t)NNODES * HD * 4, stream);

        cheb_scatter_kernel<<<gE, TB, 0, stream>>>(src, dst, norm_c, x, p, tx1);
        node_transform_kernel<<<gNH, TB, 0, stream>>>(x, p, tx1, cheb_W0, cheb_W1,
                                                      cheb_b, Wz, Wr, Wh,
                                                      hbuf, xz, xr, xh);
        gcn_scatter_kernel<<<gScat, TB, 0, stream>>>(src, dst, norm_g, xz, cz);
        gcn_scatter_kernel<<<gScat, TB, 0, stream>>>(src, dst, norm_g, xr, cr);
        gcn_scatter_kernel<<<gScat, TB, 0, stream>>>(src, dst, norm_g, xh, ch);
        gcn_self_kernel<<<gNH, TB, 0, stream>>>(self_g, xz, xr, xh, bz, br, bh, cz, cr, ch);

        // Z = sigmoid([cz, h] @ Lz + blz)    (Z overwrites xz)
        pack2_kernel<<<gPack, TB, 0, stream>>>(cz, hbuf, Acat);
        wmma_gemm_lds_kernel<2 * HD, HD, 1, true><<<gemm_grid(NNODES, HD), TB, 0, stream>>>(
            Acat, LzT, blz, xz, NNODES);
        // R = sigmoid([cr, h] @ Lr + blr)    (R overwrites xr)
        pack2_kernel<<<gPack, TB, 0, stream>>>(cr, hbuf, Acat);
        wmma_gemm_lds_kernel<2 * HD, HD, 1, true><<<gemm_grid(NNODES, HD), TB, 0, stream>>>(
            Acat, LrT, blr, xr, NNODES);
        // Ht = tanh([ch, h*R] @ Lh + blh)    (Ht overwrites xh)
        pack2_mul_kernel<<<gPack, TB, 0, stream>>>(ch, hbuf, xr, Acat);
        wmma_gemm_lds_kernel<2 * HD, HD, 2, true><<<gemm_grid(NNODES, HD), TB, 0, stream>>>(
            Acat, LhT, blh, xh, NNODES);

        gru_update_kernel<<<gNH, TB, 0, stream>>>(xz, hbuf, xh, probs, p, Hacc);
    }

    // ---- head MLP ----
    relu_pack_kernel<<<gNH, TB, 0, stream>>>(Hacc, Hr);
    wmma_gemm_lds_kernel<HD, HID, 3, true><<<gemm_grid(NNODES, HID), TB, 0, stream>>>(
        Hr, W1T, b1, h1, NNODES);
    head2_kernel<<<(NNODES * ODIM + TB - 1) / TB, TB, 0, stream>>>(h1, W2, b2, out);

    // second tuple output: out_hidden = Hacc
    (void)hipMemcpyAsync(out + (size_t)NNODES * ODIM, Hacc, (size_t)NNODES * HD * 4,
                         hipMemcpyDeviceToDevice, stream);
}
